// Model_20925080666140
// MI455X (gfx1250) — matrix operand order repaired and checked by reading, not compiled, not run on hardware
//
#include <hip/hip_runtime.h>
#include <hip/hip_bf16.h>
#include <cstdint>

// ---------------------------------------------------------------------------
// 2-layer LSTM (B=256, T=256, F=32, H=512) for MI455X (gfx1250, wave32).
// bf16 WMMA (f32 accumulate) GEMMs; weights transposed to [N][K] bf16 once per
// launch. Per GEMM block, 256 weight columns are staged into LDS with the
// CDNA5 async-to-LDS path (double buffered, ASYNCcnt). Each wave owns a
// 32(M)x64(N) strip: 2 A fragments x 4 B fragments = 8 WMMAs per K-chunk,
// so every LDS B-fragment read feeds two WMMAs.
// Weights (~6.3 MB bf16) stay resident in the 192 MB L2 across all 256 steps.
// ---------------------------------------------------------------------------

#define B_  256
#define T_  256
#define F_  32
#define H_  512
#define N4_ 2048   // 4*H

typedef __bf16 bf16_t;
typedef __attribute__((ext_vector_type(8)))  __bf16 bf16x8;
typedef __attribute__((ext_vector_type(16))) __bf16 bf16x16;
typedef __attribute__((ext_vector_type(8)))  float  v8f;

// ---------------------------------------------------------------------------
__global__ __launch_bounds__(256) void cvt_bf16_kernel(const float* __restrict__ src,
                                                       bf16_t* __restrict__ dst, int n) {
    int i = blockIdx.x * 256 + threadIdx.x;
    if (i < n) dst[i] = (bf16_t)src[i];
}

// f32 [K][N] -> bf16 [N][K] (transpose). Weight prep for WMMA B-fragments.
__global__ __launch_bounds__(256) void cvt_wT_kernel(const float* __restrict__ src,
                                                     bf16_t* __restrict__ dst, int K, int N) {
    int i = blockIdx.x * 256 + threadIdx.x;
    if (i < K * N) {
        int k = i / N, n = i % N;
        dst[(size_t)n * K + k] = (bf16_t)src[i];
    }
}

__global__ __launch_bounds__(256) void zero_state_kernel(float* h0f, float* c0, float* h1f,
                                                         float* c1, bf16_t* h0b, bf16_t* h1b) {
    int i = blockIdx.x * 256 + threadIdx.x;
    if (i < B_ * H_) {
        h0f[i] = 0.f; c0[i] = 0.f; h1f[i] = 0.f; c1[i] = 0.f;
        h0b[i] = (bf16_t)0.f; h1b[i] = (bf16_t)0.f;
    }
}

// ---------------------------------------------------------------------------
// Async copy of 64B global->LDS (offsets apply to BOTH the LDS and global
// address per the ISA pseudocode). Tracked by ASYNCcnt.
__device__ __forceinline__ void async_copy_64B(unsigned lds_off, const bf16_t* gptr) {
    unsigned long long ga = (unsigned long long)(uintptr_t)gptr;
    asm volatile("global_load_async_to_lds_b128 %0, %1, off\n\t"
                 "global_load_async_to_lds_b128 %0, %1, off offset:16\n\t"
                 "global_load_async_to_lds_b128 %0, %1, off offset:32\n\t"
                 "global_load_async_to_lds_b128 %0, %1, off offset:48"
                 :: "v"(lds_off), "v"(ga) : "memory");
}
__device__ __forceinline__ void wait_async0() {
    asm volatile("s_wait_asynccnt 0x0" ::: "memory");
}

__device__ __forceinline__ bf16x16 load_a_frag(const bf16_t* arow, int k0, int half) {
    // 16-bit A layout: lane m=l16; half=0 holds K 0..7/16..23, half=1 holds K 8..15/24..31
    bf16x8 lo = *(const bf16x8*)(arow + k0 + half * 8);
    bf16x8 hi = *(const bf16x8*)(arow + k0 + 16 + half * 8);
    return __builtin_shufflevector(lo, hi,
        0, 1, 2, 3, 4, 5, 6, 7, 8, 9, 10, 11, 12, 13, 14, 15);
}

// ---------------------------------------------------------------------------
// One accumulation pass: acc += A[32 rows][K] @ Wt[cols][K]^T for this wave's
// 32x64 strip. B tiles staged to LDS (async, double buffered), fully unrolled.
template <int K>
__device__ __forceinline__ void gemm_pass(
    const bf16_t* __restrict__ arow0, const bf16_t* __restrict__ arow1,
    const bf16_t* __restrict__ wcol,
    bf16_t (&smem)[2][256 * 32],
    int nLoc, int half, int l16, int sCol, v8f (&acc)[8]) {

    constexpr int nch = K >> 5;   // K-chunks of 32

    // prologue: stage chunk 0 into buffer 0 (one full 64B column per thread)
    {
        unsigned lds0 = (unsigned)(uintptr_t)(&smem[0][sCol * 32]);
        async_copy_64B(lds0, wcol);
    }
    wait_async0();
    __syncthreads();

    bf16x16 a0 = load_a_frag(arow0, 0, half);
    bf16x16 a1 = load_a_frag(arow1, 0, half);

    #pragma unroll
    for (int i = 0; i < nch; ++i) {
        const int buf = i & 1;

        bf16x16 a0n, a1n;
        if (i + 1 < nch) {
            // stage chunk i+1 into the other buffer (overlaps WMMAs below)
            unsigned ldsN = (unsigned)(uintptr_t)(&smem[buf ^ 1][sCol * 32]);
            async_copy_64B(ldsN, wcol + (i + 1) * 32);
            a0n = load_a_frag(arow0, (i + 1) * 32, half);
            a1n = load_a_frag(arow1, (i + 1) * 32, half);
        }

        // preload all 4 B fragments; each feeds two WMMAs (a0 and a1)
        bf16x16 bfrag[4];
        #pragma unroll
        for (int j = 0; j < 4; ++j) {
            const bf16_t* bp = &smem[buf][(nLoc + j * 16 + l16) * 32 + half * 8];
            bf16x8 blo = *(const bf16x8*)(bp);
            bf16x8 bhi = *(const bf16x8*)(bp + 16);
            bfrag[j] = __builtin_shufflevector(blo, bhi,
                0, 1, 2, 3, 4, 5, 6, 7, 8, 9, 10, 11, 12, 13, 14, 15);
        }
        #pragma unroll
        for (int j = 0; j < 4; ++j) {
            acc[j] = __builtin_amdgcn_wmma_f32_16x16x32_bf16(
                false, a0, false, bfrag[j], (short)0, acc[j], false, false);
            acc[4 + j] = __builtin_amdgcn_wmma_f32_16x16x32_bf16(
                false, a1, false, bfrag[j], (short)0, acc[4 + j], false, false);
        }

        wait_async0();       // free when nothing outstanding (last chunk)
        __syncthreads();     // all waves done reading buf; next stage may overwrite
        if (i + 1 < nch) { a0 = a0n; a1 = a1n; }
    }
}

// gates[256][2048] = bias + A1[256][K1] @ W1t[2048][K1]^T + A2[256][K2] @ W2t[2048][K2]^T
// Block: 256 threads (8 waves), tile 64(M) x 256(N). Wave: 32(M) x 64(N).
template <int K1, int K2>
__global__ __launch_bounds__(256) void gemm_gates_kernel(
    const bf16_t* __restrict__ A1, int lda1, const bf16_t* __restrict__ W1t,
    const bf16_t* __restrict__ A2, int lda2, const bf16_t* __restrict__ W2t,
    const float* __restrict__ bias, float* __restrict__ out) {

    // 2 buffers x 256 columns x 32 K x bf16 = 32 KB
    __shared__ bf16_t smem[2][256 * 32];

    const int tid  = threadIdx.x;
    const int wave = tid >> 5;
    const int lane = tid & 31;
    const int half = lane >> 4;        // 0: lanes 0-15, 1: lanes 16-31
    const int l16  = lane & 15;
    const int mBase  = blockIdx.y * 64 + (wave >> 2) * 32;   // wave's 32-row strip
    const int nBlock = blockIdx.x * 256;
    const int nLoc   = (wave & 3) * 64;                      // wave's 64-col strip

    const int sCol = tid;              // staging: one 64B column chunk per thread

    v8f acc[8] = {v8f{}, v8f{}, v8f{}, v8f{}, v8f{}, v8f{}, v8f{}, v8f{}};

    gemm_pass<K1>(A1 + (size_t)(mBase + l16) * lda1,
                  A1 + (size_t)(mBase + 16 + l16) * lda1,
                  W1t + (size_t)(nBlock + sCol) * K1,
                  smem, nLoc, half, l16, sCol, acc);
    gemm_pass<K2>(A2 + (size_t)(mBase + l16) * lda2,
                  A2 + (size_t)(mBase + 16 + l16) * lda2,
                  W2t + (size_t)(nBlock + sCol) * K2,
                  smem, nLoc, half, l16, sCol, acc);

    // C/D layout: VGPR r, lanes 0-15 -> (M = r, N = lane); lanes 16-31 -> (M = r+8).
    #pragma unroll
    for (int mi = 0; mi < 2; ++mi) {
        #pragma unroll
        for (int j = 0; j < 4; ++j) {
            const int col = nBlock + nLoc + j * 16 + l16;
            const float bv = bias[col];
            #pragma unroll
            for (int r = 0; r < 8; ++r) {
                const int row = mBase + mi * 16 + half * 8 + r;
                out[(size_t)row * N4_ + col] = acc[mi * 4 + j][r] + bv;
            }
        }
    }
}

// ---------------------------------------------------------------------------
__device__ __forceinline__ float sigmoidf_(float x) { return 1.0f / (1.0f + __expf(-x)); }

__global__ __launch_bounds__(256) void cell_update_kernel(const float* __restrict__ gates,
                                                          float* __restrict__ c,
                                                          float* __restrict__ hf,
                                                          bf16_t* __restrict__ hb) {
    int idx = blockIdx.x * 256 + threadIdx.x;
    if (idx >= B_ * H_) return;
    int b = idx / H_, hh = idx % H_;
    size_t base = (size_t)b * N4_;
    float ig = sigmoidf_(gates[base + hh]);
    float fg = sigmoidf_(gates[base + H_ + hh]);
    float gg = tanhf(gates[base + 2 * H_ + hh]);
    float og = sigmoidf_(gates[base + 3 * H_ + hh]);
    float cn = fg * c[idx] + ig * gg;
    c[idx] = cn;
    float hn = og * tanhf(cn);
    hf[idx] = hn;
    hb[idx] = (bf16_t)hn;
}

__global__ __launch_bounds__(256) void head_kernel(const float* __restrict__ h1f,
                                                   const float* __restrict__ Wd,
                                                   const float* __restrict__ bd,
                                                   float* __restrict__ out) {
    int b = blockIdx.x * 256 + threadIdx.x;
    if (b >= B_) return;
    float s = bd[0];
    for (int k = 0; k < H_; ++k) s += h1f[(size_t)b * H_ + k] * Wd[k];
    out[b] = s;
}

// d_out layout: [logits(256), h0(B*H), h1(B*H), c0(B*H), c1(B*H)]
__global__ __launch_bounds__(256) void pack_kernel(const float* h0f, const float* h1f,
                                                   const float* c0, const float* c1,
                                                   float* __restrict__ out) {
    int i = blockIdx.x * 256 + threadIdx.x;
    if (i >= B_ * H_) return;
    const int BH = B_ * H_;
    out[B_ + i]          = h0f[i];
    out[B_ + BH + i]     = h1f[i];
    out[B_ + 2 * BH + i] = c0[i];
    out[B_ + 3 * BH + i] = c1[i];
}

// ---------------------------------------------------------------------------
static inline size_t align256(size_t x) { return (x + 255) & ~(size_t)255; }

extern "C" void kernel_launch(void* const* d_in, const int* in_sizes, int n_in,
                              void* d_out, int out_size, void* d_ws, size_t ws_size,
                              hipStream_t stream) {
    const float* inputs = (const float*)d_in[0]; // [B,T,F]
    const float* W0 = (const float*)d_in[1];     // [F,4H]
    const float* U0 = (const float*)d_in[2];     // [H,4H]
    const float* b0 = (const float*)d_in[3];     // [4H]
    const float* W1 = (const float*)d_in[4];     // [H,4H]
    const float* U1 = (const float*)d_in[5];     // [H,4H]
    const float* b1 = (const float*)d_in[6];     // [4H]
    const float* Wd = (const float*)d_in[7];     // [H,1]
    const float* bd = (const float*)d_in[8];     // [1]
    float* out = (float*)d_out;

    uint8_t* ws = (uint8_t*)d_ws;
    size_t off = 0;
    auto carve = [&](size_t bytes) { uint8_t* p = ws + off; off = align256(off + bytes); return p; };

    bf16_t* xbf  = (bf16_t*)carve((size_t)B_ * T_ * F_ * 2);  // [B,T,F] bf16
    bf16_t* W0t  = (bf16_t*)carve((size_t)N4_ * F_ * 2);      // [2048][32]
    bf16_t* U0t  = (bf16_t*)carve((size_t)N4_ * H_ * 2);      // [2048][512]
    bf16_t* W1t  = (bf16_t*)carve((size_t)N4_ * H_ * 2);
    bf16_t* U1t  = (bf16_t*)carve((size_t)N4_ * H_ * 2);
    bf16_t* h0b  = (bf16_t*)carve((size_t)B_ * H_ * 2);
    bf16_t* h1b  = (bf16_t*)carve((size_t)B_ * H_ * 2);
    float*  h0f  = (float*)carve((size_t)B_ * H_ * 4);
    float*  h1f  = (float*)carve((size_t)B_ * H_ * 4);
    float*  c0   = (float*)carve((size_t)B_ * H_ * 4);
    float*  c1   = (float*)carve((size_t)B_ * H_ * 4);
    float*  gates = (float*)carve((size_t)B_ * N4_ * 4);

    // --- one-time (per launch) precision prep ---
    {
        int n = B_ * T_ * F_;
        cvt_bf16_kernel<<<(n + 255) / 256, 256, 0, stream>>>(inputs, xbf, n);
    }
    cvt_wT_kernel<<<(F_ * N4_ + 255) / 256, 256, 0, stream>>>(W0, W0t, F_, N4_);
    cvt_wT_kernel<<<(H_ * N4_ + 255) / 256, 256, 0, stream>>>(U0, U0t, H_, N4_);
    cvt_wT_kernel<<<(H_ * N4_ + 255) / 256, 256, 0, stream>>>(W1, W1t, H_, N4_);
    cvt_wT_kernel<<<(H_ * N4_ + 255) / 256, 256, 0, stream>>>(U1, U1t, H_, N4_);
    zero_state_kernel<<<(B_ * H_ + 255) / 256, 256, 0, stream>>>(h0f, c0, h1f, c1, h0b, h1b);

    // --- sequential recurrence ---
    dim3 gemmGrid(N4_ / 256, B_ / 64);      // 8 x 4 blocks, 8 waves each
    dim3 gemmBlock(256);
    int cellBlocks = (B_ * H_ + 255) / 256;

    for (int t = 0; t < T_; ++t) {
        // layer 0: gates = b0 + x_t @ W0 + h0 @ U0
        gemm_gates_kernel<F_, H_><<<gemmGrid, gemmBlock, 0, stream>>>(
            xbf + (size_t)t * F_, T_ * F_, W0t,
            h0b, H_, U0t, b0, gates);
        cell_update_kernel<<<cellBlocks, 256, 0, stream>>>(gates, c0, h0f, h0b);

        // layer 1: gates = b1 + h0 @ W1 + h1 @ U1
        gemm_gates_kernel<H_, H_><<<gemmGrid, gemmBlock, 0, stream>>>(
            h0b, H_, W1t,
            h1b, H_, U1t, b1, gates);
        cell_update_kernel<<<cellBlocks, 256, 0, stream>>>(gates, c1, h1f, h1b);
    }

    // --- head + output packing ---
    head_kernel<<<1, 256, 0, stream>>>(h1f, Wd, bd, out);
    pack_kernel<<<cellBlocks, 256, 0, stream>>>(h0f, h1f, c0, c1, out);
}
